// BeatGenerator_54915451846779
// MI455X (gfx1250) — compile-verified, matrix operand order checked by reading, not run on hardware
//
#include <hip/hip_runtime.h>

// ---------------------------------------------------------------------------
// MI455X (gfx1250, wave32) implementation.
// Matrix path: bf16 WMMA v_wmma_f32_16x16x32_bf16 (fp32 accumulate).
// GRU kernel: one wave owns two 16x16 output tiles (32 batch rows) so each
// weight fragment load feeds two WMMAs (halves L2 weight traffic).
// ---------------------------------------------------------------------------

typedef __bf16 v16bf __attribute__((ext_vector_type(16)));
typedef __bf16 v8bf  __attribute__((ext_vector_type(8)));
typedef float  v8f   __attribute__((ext_vector_type(8)));

#define BSZ   2048
#define TT    34
#define HH    128
#define ZZ    16

static __device__ __forceinline__ __bf16 f2bf(float f) {
  union { unsigned u; float f; } uf; uf.f = f;
  unsigned r = uf.u + 0x7FFFu + ((uf.u >> 16) & 1u);   // round-to-nearest-even
  union { unsigned short s; __bf16 b; } ub; ub.s = (unsigned short)(r >> 16);
  return ub.b;
}
static __device__ __forceinline__ float bf2f(__bf16 h) {
  union { unsigned short s; __bf16 b; } ub; ub.b = h;
  union { unsigned u; float f; } uf; uf.u = ((unsigned)ub.s) << 16;
  return uf.f;
}

// A/B fragment for 16x16x32 bf16 WMMA.
// Lane L<16 : row/col = L,    K chunks [0..7]  and [16..23]
// Lane L>=16: row/col = L-16, K chunks [8..15] and [24..31]
// Caller passes p = base + row*ld + koff (koff = 0 or 8).
static __device__ __forceinline__ v16bf load_frag(const __bf16* __restrict__ p) {
  v8bf lo = *(const v8bf*)(p);
  v8bf hi = *(const v8bf*)(p + 16);
  v16bf f;
#pragma unroll
  for (int i = 0; i < 8; ++i) { f[i] = lo[i]; f[i + 8] = hi[i]; }
  return f;
}
// Unaligned variant (window-conv rows start at 60B strides).
static __device__ __forceinline__ v16bf load_frag_u(const __bf16* __restrict__ p) {
  v16bf f;
#pragma unroll
  for (int i = 0; i < 8; ++i) { f[i] = p[i]; f[i + 8] = p[i + 16]; }
  return f;
}

static __device__ __forceinline__ float sigmoidf(float x) {
  return 1.f / (1.f + __expf(-x));
}

// ---------------------------------------------------------------------------
// fp32 -> bf16 convert with zero padding:
// dst (Ndst x Kdst), src (Nsrc x Ksrc) placed at column `shift`.
// ---------------------------------------------------------------------------
__global__ void cvt_pad_kernel(__bf16* __restrict__ dst, const float* __restrict__ src,
                               int Ndst, int Nsrc, int Kdst, int Ksrc, int shift,
                               unsigned long long tot) {
  unsigned long long i = (unsigned long long)blockIdx.x * 256 + threadIdx.x;
  if (i >= tot) return;
  int k = (int)(i % Kdst);
  int n = (int)(i / Kdst);
  float v = 0.f;
  if (n < Nsrc && k >= shift && k < shift + Ksrc)
    v = src[(size_t)n * Ksrc + (k - shift)];
  dst[i] = f2bf(v);
}

// ---------------------------------------------------------------------------
// Speaker latent path (tiny: 16-wide GEMMs, plain VALU).
// ---------------------------------------------------------------------------
__global__ void spk_kernel(const float* __restrict__ spk_table, const int* __restrict__ vid,
                           const float* __restrict__ slw, const float* __restrict__ slb,
                           const float* __restrict__ muw, const float* __restrict__ mub,
                           const float* __restrict__ lvw, const float* __restrict__ lvb,
                           const float* __restrict__ eps,
                           float* __restrict__ zmu, float* __restrict__ zlv,
                           float* __restrict__ zctx_out, float* __restrict__ zctx_ws) {
  __shared__ float zc0[16][17];
  const int j  = threadIdx.x;            // 0..15 output feature
  const int yr = threadIdx.y;            // 0..15 local row
  const int b  = blockIdx.x * 16 + yr;
  const float* emb = spk_table + (size_t)vid[b] * ZZ;
  float a = slb[j];
#pragma unroll
  for (int k = 0; k < ZZ; ++k) a += emb[k] * slw[j * ZZ + k];
  zc0[yr][j] = a;
  __syncthreads();
  float mu = mub[j], lv = lvb[j];
#pragma unroll
  for (int k = 0; k < ZZ; ++k) {
    float z = zc0[yr][k];
    mu += z * muw[j * ZZ + k];
    lv += z * lvw[j * ZZ + k];
  }
  float zc = mu + eps[b * ZZ + j] * __expf(0.5f * lv);
  zmu[b * ZZ + j] = mu;
  zlv[b * ZZ + j] = lv;
  zctx_out[b * ZZ + j] = zc;
  zctx_ws[b * ZZ + j] = zc;
}

// ---------------------------------------------------------------------------
// Window conv: xs[t,b,16+j] = win0 @ squ_w.T + win1 @ oenv_w.T + biases.
// A rows gathered from padded beats (row length 1088, data at [30,1050)).
// grid = (4352 Mtiles, 8 Ntiles), block = 32 (one wave).
// ---------------------------------------------------------------------------
__global__ void __launch_bounds__(32)
winconv_kernel(const __bf16* __restrict__ pbeats,     // (2048, 2, 1088)
               const __bf16* __restrict__ squ_wb,     // (128, 96)
               const __bf16* __restrict__ oenv_wb,    // (128, 96)
               const float* __restrict__ squ_b, const float* __restrict__ oenv_b,
               __bf16* __restrict__ xs) {             // (34, 2048, 160)
  const int lane = threadIdx.x;
  const int m0 = blockIdx.x * 16, n0 = blockIdx.y * 16;
  const int row = lane & 15;
  const int koff = (lane < 16) ? 0 : 8;
  const int m = m0 + row;
  const int b = m / TT, t = m % TT;
  const int j = n0 + row;
  const __bf16* a0 = pbeats + ((size_t)b * 2 + 0) * 1088 + t * 30 + koff;
  const __bf16* a1 = pbeats + ((size_t)b * 2 + 1) * 1088 + t * 30 + koff;
  const __bf16* w0 = squ_wb  + (size_t)j * 96 + koff;
  const __bf16* w1 = oenv_wb + (size_t)j * 96 + koff;
  float bv = squ_b[j] + oenv_b[j];
  v8f acc;
#pragma unroll
  for (int i = 0; i < 8; ++i) acc[i] = bv;
#pragma unroll
  for (int k0 = 0; k0 < 96; k0 += 32) {
    v16bf af = load_frag_u(a0 + k0);
    v16bf wf = load_frag(w0 + k0);
    acc = __builtin_amdgcn_wmma_f32_16x16x32_bf16(false, af, false, wf, (short)0, acc, false, false);
    af = load_frag_u(a1 + k0);
    wf = load_frag(w1 + k0);
    acc = __builtin_amdgcn_wmma_f32_16x16x32_bf16(false, af, false, wf, (short)0, acc, false, false);
  }
#pragma unroll
  for (int r = 0; r < 8; ++r) {
    int mm = m0 + r + ((lane < 16) ? 0 : 8);
    int bb = mm / TT, tt = mm % TT;
    xs[((size_t)tt * BSZ + bb) * 160 + 16 + n0 + row] = f2bf(acc[r]);
  }
}

// ---------------------------------------------------------------------------
// Generic bf16 WMMA GEMM: out = A(M x K) @ W(N x K).T + bias.
// remap: 0 none; 1: m=(b*34+t) -> orow=t*2048+b; 2: m=(t*2048+b) -> orow=b*34+t
// ---------------------------------------------------------------------------
__global__ void __launch_bounds__(32)
gemm_bias_kernel(const __bf16* __restrict__ A, int lda,
                 const __bf16* __restrict__ W, int ldw,
                 const float* __restrict__ bias, int nvalid, int K,
                 __bf16* __restrict__ outb, float* __restrict__ outf,
                 int ldo, int col_off, int remap) {
  const int lane = threadIdx.x;
  const int m0 = blockIdx.x * 16, n0 = blockIdx.y * 16;
  const int row = lane & 15;
  const int koff = (lane < 16) ? 0 : 8;
  const int j = n0 + row;
  float bv = (j < nvalid) ? bias[j] : 0.f;
  v8f acc;
#pragma unroll
  for (int i = 0; i < 8; ++i) acc[i] = bv;
  const __bf16* arow = A + (size_t)(m0 + row) * lda + koff;
  const __bf16* wrow = W + (size_t)j * ldw + koff;
  for (int k0 = 0; k0 < K; k0 += 32) {
    v16bf af = load_frag(arow + k0);
    v16bf wf = load_frag(wrow + k0);
    acc = __builtin_amdgcn_wmma_f32_16x16x32_bf16(false, af, false, wf, (short)0, acc, false, false);
  }
  if (j >= nvalid) return;   // after all WMMAs (EXEC must be all-1 during WMMA)
#pragma unroll
  for (int r = 0; r < 8; ++r) {
    int m = m0 + r + ((lane < 16) ? 0 : 8);
    int orow = m;
    if (remap == 1)      { int b = m / TT,  t = m % TT;  orow = t * BSZ + b; }
    else if (remap == 2) { int t = m / BSZ, b = m % BSZ; orow = b * TT + t; }
    size_t oi = (size_t)orow * ldo + col_off + j;
    if (outb) outb[oi] = f2bf(acc[r]);
    else      outf[oi] = acc[r];
  }
}

// ---------------------------------------------------------------------------
// Fused GRU cell, both directions, one (t, layer) per launch.
// grid = (64 Mblocks of 32 rows, 8 Ntiles, 2 dirs), block = 32 (one wave).
// Each wave owns TWO 16x16 tiles stacked in M so every weight fragment
// feeds two WMMAs. 12 fp32 accumulators (ir/iz/in/hr/hz/hn x 2 M-halves).
// r = sig(ir+hr), z = sig(iz+hz), n = tanh(in + r*hn), h' = (1-z)n + z h.
// Writes fp32 h-state (ping-pong), bf16 h-state, and bf16 next-layer input.
// ---------------------------------------------------------------------------
__global__ void __launch_bounds__(32)
gru_cell_kernel(const __bf16* __restrict__ x, int Kx,
                const __bf16* __restrict__ wih,    // (2, 384, Kx)
                const float*  __restrict__ bih,    // (2, 384)
                const __bf16* __restrict__ whh,    // (2, 384, 128)
                const float*  __restrict__ bhh,    // (2, 384)
                const float*  __restrict__ h_in,   // (2, 2048, 128)
                const __bf16* __restrict__ hb_in,  // (2, 2048, 128)
                float*  __restrict__ h_out,
                __bf16* __restrict__ hb_out,
                __bf16* __restrict__ xnext, int xld, unsigned long long xdoff) {
  const int lane = threadIdx.x;
  const int m0 = blockIdx.x * 32;          // two M-tiles per wave
  const int n0 = blockIdx.y * 16;
  const int dir = blockIdx.z;
  const int row = lane & 15;
  const int koff = (lane < 16) ? 0 : 8;
  const int col = n0 + row;
  const size_t HSZ = (size_t)BSZ * HH;

  v8f acc[2][3], hacc[2][3];
#pragma unroll
  for (int g = 0; g < 3; ++g) {
    float bi = bih[dir * 384 + g * HH + col];
    float bh = bhh[dir * 384 + g * HH + col];
#pragma unroll
    for (int i = 0; i < 8; ++i) {
      acc[0][g][i] = bi;  acc[1][g][i] = bi;
      hacc[0][g][i] = bh; hacc[1][g][i] = bh;
    }
  }

  // gi = x @ wih.T   (K = Kx, zero-padded)
  const __bf16* arow0 = x + (size_t)(m0 + row) * Kx + koff;
  const __bf16* arow1 = x + (size_t)(m0 + 16 + row) * Kx + koff;
  for (int k0 = 0; k0 < Kx; k0 += 32) {
    v16bf a0 = load_frag(arow0 + k0);
    v16bf a1 = load_frag(arow1 + k0);
#pragma unroll
    for (int g = 0; g < 3; ++g) {
      const __bf16* wrow = wih + ((size_t)dir * 384 + g * HH + col) * Kx + koff + k0;
      v16bf wf = load_frag(wrow);
      acc[0][g] = __builtin_amdgcn_wmma_f32_16x16x32_bf16(false, a0, false, wf, (short)0, acc[0][g], false, false);
      acc[1][g] = __builtin_amdgcn_wmma_f32_16x16x32_bf16(false, a1, false, wf, (short)0, acc[1][g], false, false);
    }
  }
  // gh = h @ whh.T   (K = 128)
  const __bf16* hrow0 = hb_in + dir * HSZ + (size_t)(m0 + row) * HH + koff;
  const __bf16* hrow1 = hb_in + dir * HSZ + (size_t)(m0 + 16 + row) * HH + koff;
#pragma unroll
  for (int k0 = 0; k0 < HH; k0 += 32) {
    v16bf a0 = load_frag(hrow0 + k0);
    v16bf a1 = load_frag(hrow1 + k0);
#pragma unroll
    for (int g = 0; g < 3; ++g) {
      const __bf16* wrow = whh + ((size_t)dir * 384 + g * HH + col) * HH + koff + k0;
      v16bf wf = load_frag(wrow);
      hacc[0][g] = __builtin_amdgcn_wmma_f32_16x16x32_bf16(false, a0, false, wf, (short)0, hacc[0][g], false, false);
      hacc[1][g] = __builtin_amdgcn_wmma_f32_16x16x32_bf16(false, a1, false, wf, (short)0, hacc[1][g], false, false);
    }
  }

#pragma unroll
  for (int half = 0; half < 2; ++half) {
#pragma unroll
    for (int r = 0; r < 8; ++r) {
      int m = m0 + half * 16 + r + ((lane < 16) ? 0 : 8);
      float ir = acc[half][0][r],  iz = acc[half][1][r],  in = acc[half][2][r];
      float hr = hacc[half][0][r], hz = hacc[half][1][r], hn = hacc[half][2][r];
      float rg = sigmoidf(ir + hr);
      float zg = sigmoidf(iz + hz);
      float ng = tanhf(in + rg * hn);
      float ho = h_in[dir * HSZ + (size_t)m * HH + col];
      float hw = (1.f - zg) * ng + zg * ho;
      h_out[dir * HSZ + (size_t)m * HH + col] = hw;
      __bf16 hb = f2bf(hw);
      hb_out[dir * HSZ + (size_t)m * HH + col] = hb;
      xnext[(size_t)m * xld + (size_t)dir * xdoff + col] = hb;
    }
  }
}

// ---------------------------------------------------------------------------
// featin[t,b,:] = [ lastf+lastb (128) | z_context (16) | zeros (16) ] (bf16)
// ---------------------------------------------------------------------------
__global__ void featpack_kernel(const __bf16* __restrict__ lastf,
                                const __bf16* __restrict__ lastb,
                                const float* __restrict__ zctx,
                                __bf16* __restrict__ featin,
                                unsigned long long tot) {
  unsigned long long i = (unsigned long long)blockIdx.x * 256 + threadIdx.x;
  if (i >= tot) return;
  int c = (int)(i % 160);
  size_t rb = (size_t)(i / 160);       // rb = t*2048 + b
  int b = (int)(rb % BSZ);
  float v;
  if (c < HH)       v = bf2f(lastf[rb * HH + c]) + bf2f(lastb[rb * HH + c]);
  else if (c < 144) v = zctx[b * ZZ + (c - HH)];
  else              v = 0.f;
  featin[i] = f2bf(v);
}

// ---------------------------------------------------------------------------
extern "C" void kernel_launch(void* const* d_in, const int* in_sizes, int n_in,
                              void* d_out, int out_size, void* d_ws, size_t ws_size,
                              hipStream_t stream) {
  const float* pre_seq   = (const float*)d_in[0];
  const float* beats     = (const float*)d_in[1];
  const int*   vid       = (const int*)d_in[4];
  const float* eps       = (const float*)d_in[5];
  const float* spk_table = (const float*)d_in[6];
  const float* spk_lin_w = (const float*)d_in[7];
  const float* spk_lin_b = (const float*)d_in[8];
  const float* mu_w  = (const float*)d_in[9];
  const float* mu_b  = (const float*)d_in[10];
  const float* lv_w  = (const float*)d_in[11];
  const float* lv_b  = (const float*)d_in[12];
  const float* squ_w = (const float*)d_in[13];
  const float* squ_b = (const float*)d_in[14];
  const float* oenv_w = (const float*)d_in[15];
  const float* oenv_b = (const float*)d_in[16];
  const float* pose_w = (const float*)d_in[17];
  const float* pose_b = (const float*)d_in[18];
  const float* out_w  = (const float*)d_in[19];
  const float* out_b  = (const float*)d_in[20];
  const float* bo_w1  = (const float*)d_in[21];
  const float* bo_b1  = (const float*)d_in[22];
  const float* bo_w2  = (const float*)d_in[23];
  const float* bo_b2  = (const float*)d_in[24];
  const float* w_ih_l0 = (const float*)d_in[25];
  const float* w_hh_l0 = (const float*)d_in[26];
  const float* b_ih_l0 = (const float*)d_in[27];
  const float* b_hh_l0 = (const float*)d_in[28];
  const float* w_ih = (const float*)d_in[29];
  const float* w_hh = (const float*)d_in[30];
  const float* b_ih = (const float*)d_in[31];
  const float* b_hh = (const float*)d_in[32];

  // ---- workspace bump allocator (256B aligned) ----
  char* base = (char*)d_ws;
  size_t off = 0;
  auto alloc = [&](size_t bytes) -> void* {
    void* p = base + off;
    off += (bytes + 255) & ~(size_t)255;
    return p;
  };
  const size_t HSZ = (size_t)BSZ * HH;                  // 2048*128

  __bf16* pbeats = (__bf16*)alloc((size_t)BSZ * 2 * 1088 * 2);
  __bf16* pre_b  = (__bf16*)alloc((size_t)BSZ * TT * 32 * 2);
  __bf16* xs_b   = (__bf16*)alloc((size_t)TT * BSZ * 160 * 2);
  __bf16* squ_wb  = (__bf16*)alloc((size_t)128 * 96 * 2);
  __bf16* oenv_wb = (__bf16*)alloc((size_t)128 * 96 * 2);
  __bf16* pose_wb = (__bf16*)alloc((size_t)16 * 32 * 2);
  __bf16* wih0b = (__bf16*)alloc((size_t)2 * 384 * 160 * 2);
  __bf16* whh0b = (__bf16*)alloc((size_t)2 * 384 * 128 * 2);
  __bf16* wihb  = (__bf16*)alloc((size_t)6 * 384 * 256 * 2);
  __bf16* whhb  = (__bf16*)alloc((size_t)6 * 384 * 128 * 2);
  __bf16* outwb = (__bf16*)alloc((size_t)128 * 160 * 2);
  __bf16* bo1b  = (__bf16*)alloc((size_t)64 * 128 * 2);
  __bf16* bo2b  = (__bf16*)alloc((size_t)32 * 64 * 2);
  float*  hA  = (float*)alloc(8 * HSZ * 4);
  float*  hB  = (float*)alloc(8 * HSZ * 4);
  __bf16* hAb = (__bf16*)alloc(8 * HSZ * 2);
  __bf16* hBb = (__bf16*)alloc(8 * HSZ * 2);
  __bf16* xc[3];
  for (int l = 0; l < 3; ++l) xc[l] = (__bf16*)alloc((size_t)BSZ * 256 * 2);
  __bf16* lastf = (__bf16*)alloc((size_t)TT * HSZ * 2);
  __bf16* lastb = (__bf16*)alloc((size_t)TT * HSZ * 2);
  float*  zctx  = (float*)alloc((size_t)BSZ * ZZ * 4);
  __bf16* featin = (__bf16*)alloc((size_t)TT * BSZ * 160 * 2);
  __bf16* featb  = (__bf16*)alloc((size_t)TT * HSZ * 2);
  __bf16* hidb   = (__bf16*)alloc((size_t)TT * BSZ * 64 * 2);

  float* out_beat = (float*)d_out;
  float* out_zctx = out_beat + (size_t)BSZ * TT * 27;
  float* out_zmu  = out_zctx + (size_t)BSZ * ZZ;
  float* out_zlv  = out_zmu  + (size_t)BSZ * ZZ;

  auto cvt = [&](__bf16* dst, const float* src, int Nd, int Ns, int Kd, int Ks, int shift) {
    unsigned long long tot = (unsigned long long)Nd * Kd;
    int blocks = (int)((tot + 255) / 256);
    cvt_pad_kernel<<<blocks, 256, 0, stream>>>(dst, src, Nd, Ns, Kd, Ks, shift, tot);
  };

  // ---- weight / input conversion to padded bf16 ----
  cvt(pbeats,  beats,   BSZ * 2,  BSZ * 2,  1088, 1020, 30);
  cvt(pre_b,   pre_seq, BSZ * TT, BSZ * TT, 32,   28,   0);
  cvt(squ_wb,  squ_w,   128, 128, 96, 90, 0);
  cvt(oenv_wb, oenv_w,  128, 128, 96, 90, 0);
  cvt(pose_wb, pose_w,  16,  16,  32, 28, 0);
  cvt(wih0b,   w_ih_l0, 2 * 384, 2 * 384, 160, 144, 0);
  cvt(whh0b,   w_hh_l0, 2 * 384, 2 * 384, 128, 128, 0);
  cvt(wihb,    w_ih,    6 * 384, 6 * 384, 256, 256, 0);
  cvt(whhb,    w_hh,    6 * 384, 6 * 384, 128, 128, 0);
  cvt(outwb,   out_w,   128, 128, 160, 144, 0);
  cvt(bo1b,    bo_w1,   64,  64,  128, 128, 0);
  cvt(bo2b,    bo_w2,   32,  27,  64,  64,  0);

  // ---- zero init: xs pad region + initial GRU state (t=0 reads hA) ----
  hipMemsetAsync(xs_b, 0, (size_t)TT * BSZ * 160 * 2, stream);
  hipMemsetAsync(hA,   0, 8 * HSZ * 4, stream);
  hipMemsetAsync(hAb,  0, 8 * HSZ * 2, stream);

  // ---- speaker latent path ----
  spk_kernel<<<BSZ / 16, dim3(16, 16), 0, stream>>>(
      spk_table, vid, spk_lin_w, spk_lin_b, mu_w, mu_b, lv_w, lv_b, eps,
      out_zmu, out_zlv, out_zctx, zctx);

  // ---- xs = concat[pose(16), squ+oenv(128)] (remapped to (T, BS, 160)) ----
  const int MT = (BSZ * TT) / 16;    // 4352 row tiles
  gemm_bias_kernel<<<dim3(MT, 1), 32, 0, stream>>>(
      pre_b, 32, pose_wb, 32, pose_b, 16, 32, xs_b, nullptr, 160, 0, /*remap=*/1);
  winconv_kernel<<<dim3(MT, 8), 32, 0, stream>>>(
      pbeats, squ_wb, oenv_wb, squ_b, oenv_b, xs_b);

  // ---- bidirectional 4-layer GRU over T=34 steps ----
  for (int t = 0; t < TT; ++t) {
    const float*  hin_f  = (t & 1) ? hB  : hA;
    float*        hout_f = (t & 1) ? hA  : hB;
    const __bf16* hin_b  = (t & 1) ? hBb : hAb;
    __bf16*       hout_b = (t & 1) ? hAb : hBb;
    for (int l = 0; l < 4; ++l) {
      const __bf16* x   = (l == 0) ? xs_b + (size_t)t * BSZ * 160 : xc[l - 1];
      int Kx            = (l == 0) ? 160 : 256;
      const __bf16* wi  = (l == 0) ? wih0b : wihb + (size_t)(l - 1) * 2 * 384 * 256;
      const __bf16* wh  = (l == 0) ? whh0b : whhb + (size_t)(l - 1) * 2 * 384 * 128;
      const float*  bi  = (l == 0) ? b_ih_l0 : b_ih + (size_t)(l - 1) * 2 * 384;
      const float*  bh  = (l == 0) ? b_hh_l0 : b_hh + (size_t)(l - 1) * 2 * 384;
      __bf16* xnext; int xld; unsigned long long xdoff;
      if (l < 3) { xnext = xc[l]; xld = 256; xdoff = 128; }
      else { xnext = lastf + (size_t)t * HSZ; xld = 128;
             xdoff = (unsigned long long)(lastb - lastf); }
      gru_cell_kernel<<<dim3(BSZ / 32, HH / 16, 2), 32, 0, stream>>>(
          x, Kx, wi, bi, wh, bh,
          hin_f + (size_t)(2 * l) * HSZ, hin_b + (size_t)(2 * l) * HSZ,
          hout_f + (size_t)(2 * l) * HSZ, hout_b + (size_t)(2 * l) * HSZ,
          xnext, xld, xdoff);
    }
  }

  // ---- output head ----
  {
    unsigned long long tot = (unsigned long long)TT * BSZ * 160;
    featpack_kernel<<<(int)((tot + 255) / 256), 256, 0, stream>>>(
        lastf, lastb, zctx, featin, tot);
  }
  // feat = featin(69632 x 160) @ out_w.T -> bf16 (ld 128)
  gemm_bias_kernel<<<dim3(MT, 8), 32, 0, stream>>>(
      featin, 160, outwb, 160, out_b, 128, 160, featb, nullptr, 128, 0, 0);
  // hid = feat @ bo_w1.T -> bf16 (ld 64)
  gemm_bias_kernel<<<dim3(MT, 4), 32, 0, stream>>>(
      featb, 128, bo1b, 128, bo_b1, 64, 128, hidb, nullptr, 64, 0, 0);
  // beat = hid @ bo_w2.T -> fp32 d_out, remapped to (BS, T, 27), cols<27
  gemm_bias_kernel<<<dim3(MT, 2), 32, 0, stream>>>(
      hidb, 64, bo2b, 64, bo_b2, 27, 64, nullptr, out_beat, 27, 0, /*remap=*/2);
}
